// MeshGraphNetDelta_46883863003528
// MI455X (gfx1250) — matrix operand order repaired and checked by reading, not compiled, not run on hardware
//
#include <hip/hip_runtime.h>
#include <hip/hip_bf16.h>

// ---------------------------------------------------------------------------
// MeshGraphNet on gfx1250: bf16 activations, V_WMMA_F32_16X16X32_BF16 GEMMs,
// fp32 accumulate / bias / SiLU / residual, fp32 atomic scatter-add.
// Edge kernel: 64 edges/block, weight B-fragments in registers, staging via
// GLOBAL_LOAD_ASYNC_TO_LDS_B128 (ASYNCcnt), NT stores for the edge stream.
// ---------------------------------------------------------------------------

#define NN 50000
#define EE 600000
#define HH 128
#define NBLK 15

typedef __attribute__((ext_vector_type(16))) __bf16          v16bf;
typedef __attribute__((ext_vector_type(2)))  __bf16          v2bf;
typedef __attribute__((ext_vector_type(8)))  float           v8f;
typedef __attribute__((ext_vector_type(8)))  unsigned short  ushort8v;

union FragU { v16bf v; ushort8v h[2]; };

__device__ __forceinline__ unsigned short f2bf_bits(float f) {
    unsigned u = __builtin_bit_cast(unsigned, f);
    unsigned r = u + 0x7FFFu + ((u >> 16) & 1u);   // round-to-nearest-even
    return (unsigned short)(r >> 16);
}
// Packed pair conversion: one v_cvt_pk_bf16_f32 when available.
__device__ __forceinline__ unsigned f2bf_pair(float lo, float hi) {
#if __has_builtin(__builtin_amdgcn_cvt_pk_bf16_f32)
    v2bf p = __builtin_amdgcn_cvt_pk_bf16_f32(lo, hi);
    return __builtin_bit_cast(unsigned, p);
#else
    return (unsigned)f2bf_bits(lo) | ((unsigned)f2bf_bits(hi) << 16);
#endif
}
__device__ __forceinline__ float bf2f(unsigned short h) {
    unsigned u = ((unsigned)h) << 16;
    return __builtin_bit_cast(float, u);
}
__device__ __forceinline__ float silu_fast(float x) {
    return x * __builtin_amdgcn_rcpf(1.0f + __expf(-x));   // x * sigmoid(x)
}

__device__ __forceinline__ v8f wmma_bf16(v16bf a, v16bf b, v8f c) {
    return __builtin_amdgcn_wmma_f32_16x16x32_bf16(
        /*neg_a=*/false, a, /*neg_b=*/false, b,
        /*c_mod=*/(short)0, c, /*reuse_a=*/false, /*reuse_b=*/false);
}

// Async DMA: 16B global -> LDS, per-lane addresses, tracked by ASYNCcnt.
__device__ __forceinline__ void async_load_b128(void* lds_dst, const void* gsrc) {
    unsigned lds_off = (unsigned)(unsigned long long)
        (__attribute__((address_space(3))) char*)lds_dst;
    unsigned long long ga = (unsigned long long)gsrc;
    asm volatile("global_load_async_to_lds_b128 %0, %1, off"
                 :: "v"(lds_off), "v"(ga) : "memory");
}
__device__ __forceinline__ void wait_asynccnt0() {
    asm volatile("s_wait_asynccnt 0" ::: "memory");
}

// A-matrix 16x32 bf16 fragment from a row-major [16, K1] bf16 LDS panel.
// Lane L -> row M=L%16, kbase=(L<16)?0:8; two contiguous b128 loads.
__device__ __forceinline__ v16bf load_a_frag(const unsigned short* sA, int K1, int kt, int lane) {
    int m  = lane & 15;
    int kb = (lane & 16) >> 1;                 // 0 or 8
    const unsigned short* p = sA + m * K1 + kt * 32 + kb;
    FragU f;
    f.h[0] = *(const ushort8v*)(p);
    f.h[1] = *(const ushort8v*)(p + 16);
    return f.v;
}

// B-matrix 32x16 bf16 fragment from pre-permuted weight buffer:
// buf[((kt*8 + ntile)*32 + lane)*16 + j] holds W[kt*32 + 2*(j&8) + (lane&16)/2 + (j&7)][ntile*16 + lane%16].
__device__ __forceinline__ v16bf load_b_frag(const unsigned short* wf, int kt, int nt, int lane) {
    const unsigned short* p = wf + (((kt << 3) + nt) << 9) + (lane << 4);
    FragU f;
    f.h[0] = *(const ushort8v*)(p);
    f.h[1] = *(const ushort8v*)(p + 8);
    return f.v;
}

// Single-tile Linear->SiLU->Linear (encoder / node paths).
template <int K1>
__device__ __forceinline__ v8f mlp2_tile(const unsigned short* sA, unsigned short* sH,
                                         const unsigned short* w1f, const float* b1,
                                         const unsigned short* w2f, const float* b2) {
    const int lane = threadIdx.x & 31;
    const int wv   = threadIdx.x >> 5;         // 0..7 : output n-tile
    v8f acc = {};
#pragma unroll
    for (int kt = 0; kt < K1 / 32; ++kt)
        acc = wmma_bf16(load_a_frag(sA, K1, kt, lane), load_b_frag(w1f, kt, wv, lane), acc);

    const int ncol  = (wv << 4) + (lane & 15);
    const int rbase = (lane & 16) >> 1;        // 0 or 8
    const float bias1 = b1[ncol];
#pragma unroll
    for (int r = 0; r < 8; r += 2) {
        unsigned pk = f2bf_pair(silu_fast(acc[r] + bias1), silu_fast(acc[r + 1] + bias1));
        sH[(rbase + r)     * HH + ncol] = (unsigned short)pk;
        sH[(rbase + r + 1) * HH + ncol] = (unsigned short)(pk >> 16);
    }
    __syncthreads();

    v8f acc2 = {};
#pragma unroll
    for (int kt = 0; kt < 4; ++kt)
        acc2 = wmma_bf16(load_a_frag(sH, HH, kt, lane), load_b_frag(w2f, kt, wv, lane), acc2);
    const float bias2 = b2[ncol];
#pragma unroll
    for (int r = 0; r < 8; ++r) acc2[r] += bias2;
    return acc2;
}

// ---------------------------------------------------------------------------
// Weight fp32 -> bf16 fragment-layout conversion (zero-pads K to Kpad).
// ---------------------------------------------------------------------------
__global__ void conv_frag_kernel(const float* __restrict__ W, unsigned short* __restrict__ out,
                                 int Kpad, int Ksrc) {
    int idx = blockIdx.x * 256 + threadIdx.x;
    int total = Kpad << 7;                     // Kpad * 128
    if (idx >= total) return;
    int e    = idx & 15;
    int lane = (idx >> 4) & 31;
    int t    = idx >> 9;
    int nt   = t & 7;
    int kt   = t >> 3;
    int k = (kt << 5) + ((e & 8) << 1) + ((lane & 16) >> 1) + (e & 7);
    int n = (nt << 4) + (lane & 15);
    float v = (k < Ksrc) ? W[k * HH + n] : 0.0f;
    out[idx] = f2bf_bits(v);
}

// ---------------------------------------------------------------------------
// Encoder: [rows, IN] fp32 -> MLP2 -> [rows, 128] bf16 (IN padded to K=32).
// ---------------------------------------------------------------------------
__global__ void encoder_kernel(const float* __restrict__ in, int IN,
                               unsigned short* __restrict__ outBuf,
                               const unsigned short* __restrict__ w1f, const float* __restrict__ b1,
                               const unsigned short* __restrict__ w2f, const float* __restrict__ b2) {
    __shared__ __align__(16) unsigned short sA[16 * 32];
    __shared__ __align__(16) unsigned short sH[16 * HH];
    const int r0 = blockIdx.x << 4;
    for (int t = threadIdx.x; t < 512; t += 256) {
        int row = t >> 5, kk = t & 31;
        float v = (kk < IN) ? in[(size_t)(r0 + row) * IN + kk] : 0.0f;
        sA[t] = f2bf_bits(v);
    }
    __syncthreads();
    v8f o = mlp2_tile<32>(sA, sH, w1f, b1, w2f, b2);
    const int lane = threadIdx.x & 31, wv = threadIdx.x >> 5;
    const int ncol = (wv << 4) + (lane & 15), rbase = (lane & 16) >> 1;
#pragma unroll
    for (int r = 0; r < 8; ++r)
        outBuf[(size_t)(r0 + rbase + r) * HH + ncol] = f2bf_bits(o[r]);
}

// ---------------------------------------------------------------------------
// Edge update, 64 edges per block (4 row-tiles):
//   e_in = [e | h[src] | h[dst]] (K=384) -> MLP2 -> e_up
//   e += e_up (bf16, NT stores), agg[dst] += e_up (fp32 atomics -> L2).
// Staging uses GLOBAL_LOAD_ASYNC_TO_LDS_B128; weights reused across 4 tiles.
// ---------------------------------------------------------------------------
__global__ void edge_block_kernel(const int* __restrict__ edge_index,
                                  unsigned short* __restrict__ eBuf,
                                  const unsigned short* __restrict__ hBuf,
                                  float* __restrict__ agg,
                                  const unsigned short* __restrict__ w1f, const float* __restrict__ b1,
                                  const unsigned short* __restrict__ w2f, const float* __restrict__ b2) {
    __shared__ __align__(16) unsigned short sA[4][16 * 384];   // 48 KB
    __shared__ __align__(16) unsigned short sH[4][16 * HH];    // 16 KB
    __shared__ int sSrc[64], sDst[64];
    const int e0   = blockIdx.x << 6;          // 64 edges per block
    const int tid  = threadIdx.x;
    const int lane = tid & 31;
    const int wv   = tid >> 5;

    if (tid < 64) {
        sSrc[tid] = edge_index[e0 + tid];
        sDst[tid] = edge_index[EE + e0 + tid];
    }
    __syncthreads();

    // Stage [64, 384] bf16 panel: 3072 async 16B DMA chunks (no VGPR round trip).
    for (int c = tid; c < 3072; c += 256) {
        int row  = c / 48;                     // 0..63
        int rem  = c - row * 48;
        int seg  = rem >> 4;                   // 0=e, 1=h[src], 2=h[dst]
        int k8   = (rem & 15) << 3;
        int tile = row >> 4, r16 = row & 15;
        const unsigned short* gsrc;
        if (seg == 0)      gsrc = eBuf + (size_t)(e0 + row) * HH + k8;
        else if (seg == 1) gsrc = hBuf + (size_t)sSrc[row] * HH + k8;
        else               gsrc = hBuf + (size_t)sDst[row] * HH + k8;
        async_load_b128(sA[tile] + r16 * 384 + (seg << 7) + k8, gsrc);
    }

    // Hoist weight fragments into registers (shared across the 4 tiles).
    v16bf B1[12], B2[4];
#pragma unroll
    for (int kt = 0; kt < 12; ++kt) B1[kt] = load_b_frag(w1f, kt, wv, lane);
#pragma unroll
    for (int kt = 0; kt < 4; ++kt)  B2[kt] = load_b_frag(w2f, kt, wv, lane);

    const int ncol  = (wv << 4) + (lane & 15);
    const int rbase = (lane & 16) >> 1;
    const float bias1 = b1[ncol];
    const float bias2 = b2[ncol];
    wait_asynccnt0();                          // our DMA chunks are in LDS
    __syncthreads();                           // everyone's chunks visible

    // Layer 1 for all 4 tiles (each wave owns its 16 output columns).
#pragma unroll
    for (int t = 0; t < 4; ++t) {
        v8f acc = {};
#pragma unroll
        for (int kt = 0; kt < 12; ++kt)
            acc = wmma_bf16(load_a_frag(sA[t], 384, kt, lane), B1[kt], acc);
#pragma unroll
        for (int r = 0; r < 8; r += 2) {
            unsigned pk = f2bf_pair(silu_fast(acc[r] + bias1), silu_fast(acc[r + 1] + bias1));
            sH[t][(rbase + r)     * HH + ncol] = (unsigned short)pk;
            sH[t][(rbase + r + 1) * HH + ncol] = (unsigned short)(pk >> 16);
        }
    }
    __syncthreads();

    // Layer 2 + epilogue per tile: residual into sA seg0 (in place), scatter-add.
#pragma unroll
    for (int t = 0; t < 4; ++t) {
        v8f acc2 = {};
#pragma unroll
        for (int kt = 0; kt < 4; ++kt)
            acc2 = wmma_bf16(load_a_frag(sH[t], HH, kt, lane), B2[kt], acc2);
#pragma unroll
        for (int r = 0; r < 8; ++r) {
            int r16  = rbase + r;
            int grow = (t << 4) + r16;
            float up = acc2[r] + bias2;
            float eo = bf2f(sA[t][r16 * 384 + ncol]);
            sA[t][r16 * 384 + ncol] = f2bf_bits(eo + up);   // own column: no race
            atomicAdd(agg + (size_t)sDst[grow] * HH + ncol, up);
        }
    }
    __syncthreads();

    // Flush e_new with vectorized non-temporal 16B stores: 64 rows x 16 chunks.
    for (int c = tid; c < 1024; c += 256) {
        int row  = c >> 4;
        int tile = row >> 4, r16 = row & 15;
        int k8   = (c & 15) << 3;
        ushort8v v = *(const ushort8v*)(sA[tile] + r16 * 384 + k8);
        __builtin_nontemporal_store(v, (ushort8v*)(eBuf + (size_t)(e0 + row) * HH + k8));
    }
}

// ---------------------------------------------------------------------------
// Node update: [h | agg] (K=256) -> MLP2 -> h += x_up.
// ---------------------------------------------------------------------------
__global__ void node_block_kernel(unsigned short* __restrict__ hBuf,
                                  const float* __restrict__ agg,
                                  const unsigned short* __restrict__ w1f, const float* __restrict__ b1,
                                  const unsigned short* __restrict__ w2f, const float* __restrict__ b2) {
    __shared__ __align__(16) unsigned short sA[16 * 256];
    __shared__ __align__(16) unsigned short sH[16 * HH];
    const int n0 = blockIdx.x << 4;
    {   // h half: 256 chunks of 8 halves
        int c = threadIdx.x;
        int row = c >> 4, k8 = (c & 15) << 3;
        *(ushort8v*)(sA + row * 256 + k8) = *(const ushort8v*)(hBuf + (size_t)(n0 + row) * HH + k8);
    }
#pragma unroll
    for (int i = 0; i < 8; ++i) {              // agg half: fp32 -> bf16
        int idx = threadIdx.x + (i << 8);
        int row = idx >> 7, nn = idx & 127;
        sA[row * 256 + HH + nn] = f2bf_bits(agg[(size_t)(n0 + row) * HH + nn]);
    }
    __syncthreads();
    v8f xup = mlp2_tile<256>(sA, sH, w1f, b1, w2f, b2);
    const int lane = threadIdx.x & 31, wv = threadIdx.x >> 5;
    const int ncol = (wv << 4) + (lane & 15), rbase = (lane & 16) >> 1;
#pragma unroll
    for (int r = 0; r < 8; ++r) {
        int row = rbase + r;
        float ho = bf2f(sA[row * 256 + ncol]);
        hBuf[(size_t)(n0 + row) * HH + ncol] = f2bf_bits(ho + xup[r]);
    }
}

// ---------------------------------------------------------------------------
// Decoder: Linear(128)->SiLU (WMMA) then dot with w2[128] -> out[N] fp32.
// ---------------------------------------------------------------------------
__global__ void decoder_kernel(const unsigned short* __restrict__ hBuf,
                               const unsigned short* __restrict__ w1f, const float* __restrict__ b1,
                               const float* __restrict__ w2, const float* __restrict__ b2,
                               float* __restrict__ out) {
    __shared__ __align__(16) unsigned short sA[16 * HH];
    __shared__ float sHf[16 * HH];
    const int n0 = blockIdx.x << 4;
    {
        int c = threadIdx.x;
        int row = c >> 4, k8 = (c & 15) << 3;
        *(ushort8v*)(sA + row * HH + k8) = *(const ushort8v*)(hBuf + (size_t)(n0 + row) * HH + k8);
    }
    __syncthreads();
    const int lane = threadIdx.x & 31, wv = threadIdx.x >> 5;
    v8f acc = {};
#pragma unroll
    for (int kt = 0; kt < 4; ++kt)
        acc = wmma_bf16(load_a_frag(sA, HH, kt, lane), load_b_frag(w1f, kt, wv, lane), acc);
    const int ncol = (wv << 4) + (lane & 15), rbase = (lane & 16) >> 1;
    float bias1 = b1[ncol];
#pragma unroll
    for (int r = 0; r < 8; ++r)
        sHf[(rbase + r) * HH + ncol] = silu_fast(acc[r] + bias1);
    __syncthreads();
    if (threadIdx.x < 16) {
        float s = b2[0];
        for (int k = 0; k < HH; ++k) s += sHf[threadIdx.x * HH + k] * w2[k];
        out[n0 + threadIdx.x] = s;
    }
}

// ---------------------------------------------------------------------------
extern "C" void kernel_launch(void* const* d_in, const int* in_sizes, int n_in,
                              void* d_out, int out_size, void* d_ws, size_t ws_size,
                              hipStream_t stream) {
    const float* x           = (const float*)d_in[0];
    const float* edge_attr   = (const float*)d_in[1];
    const int*   edge_index  = (const int*)  d_in[2];
    const float* node_enc_w1 = (const float*)d_in[3];
    const float* node_enc_b1 = (const float*)d_in[4];
    const float* node_enc_w2 = (const float*)d_in[5];
    const float* node_enc_b2 = (const float*)d_in[6];
    const float* edge_enc_w1 = (const float*)d_in[7];
    const float* edge_enc_b1 = (const float*)d_in[8];
    const float* edge_enc_w2 = (const float*)d_in[9];
    const float* edge_enc_b2 = (const float*)d_in[10];
    const float* blk_edge_w1 = (const float*)d_in[11];
    const float* blk_edge_b1 = (const float*)d_in[12];
    const float* blk_edge_w2 = (const float*)d_in[13];
    const float* blk_edge_b2 = (const float*)d_in[14];
    const float* blk_node_w1 = (const float*)d_in[15];
    const float* blk_node_b1 = (const float*)d_in[16];
    const float* blk_node_w2 = (const float*)d_in[17];
    const float* blk_node_b2 = (const float*)d_in[18];
    const float* dec_w1      = (const float*)d_in[19];
    const float* dec_b1      = (const float*)d_in[20];
    const float* dec_w2      = (const float*)d_in[21];
    const float* dec_b2      = (const float*)d_in[22];

    // Workspace carve-out (256B aligned slabs).
    char*  ws  = (char*)d_ws;
    size_t off = 0;
    auto carve = [&](size_t bytes) -> void* {
        void* p = ws + off;
        off += (bytes + 255) & ~(size_t)255;
        return p;
    };
    unsigned short* hBuf   = (unsigned short*)carve((size_t)NN * HH * 2);
    unsigned short* eBuf   = (unsigned short*)carve((size_t)EE * HH * 2);
    float*          agg    = (float*)         carve((size_t)NN * HH * 4);
    unsigned short* f_nw1  = (unsigned short*)carve((size_t)32  * HH * 2);
    unsigned short* f_nw2  = (unsigned short*)carve((size_t)HH  * HH * 2);
    unsigned short* f_ew1  = (unsigned short*)carve((size_t)32  * HH * 2);
    unsigned short* f_ew2  = (unsigned short*)carve((size_t)HH  * HH * 2);
    unsigned short* f_bew1 = (unsigned short*)carve((size_t)NBLK * 384 * HH * 2);
    unsigned short* f_bew2 = (unsigned short*)carve((size_t)NBLK * HH  * HH * 2);
    unsigned short* f_bnw1 = (unsigned short*)carve((size_t)NBLK * 256 * HH * 2);
    unsigned short* f_bnw2 = (unsigned short*)carve((size_t)NBLK * HH  * HH * 2);
    unsigned short* f_dw1  = (unsigned short*)carve((size_t)HH  * HH * 2);

    auto frag = [&](const float* W, unsigned short* dst, int Kpad, int Ksrc) {
        int total = Kpad * HH;
        conv_frag_kernel<<<(total + 255) / 256, 256, 0, stream>>>(W, dst, Kpad, Ksrc);
    };
    frag(node_enc_w1, f_nw1, 32, 8);
    frag(node_enc_w2, f_nw2, HH, HH);
    frag(edge_enc_w1, f_ew1, 32, 4);
    frag(edge_enc_w2, f_ew2, HH, HH);
    for (int b = 0; b < NBLK; ++b) {
        frag(blk_edge_w1 + (size_t)b * 384 * HH, f_bew1 + (size_t)b * 384 * HH, 384, 384);
        frag(blk_edge_w2 + (size_t)b * HH  * HH, f_bew2 + (size_t)b * HH  * HH, HH, HH);
        frag(blk_node_w1 + (size_t)b * 256 * HH, f_bnw1 + (size_t)b * 256 * HH, 256, 256);
        frag(blk_node_w2 + (size_t)b * HH  * HH, f_bnw2 + (size_t)b * HH  * HH, HH, HH);
    }
    frag(dec_w1, f_dw1, HH, HH);

    // Encoders.
    encoder_kernel<<<NN / 16, 256, 0, stream>>>(x, 8, hBuf, f_nw1, node_enc_b1, f_nw2, node_enc_b2);
    encoder_kernel<<<EE / 16, 256, 0, stream>>>(edge_attr, 4, eBuf, f_ew1, edge_enc_b1, f_ew2, edge_enc_b2);

    // 15 message-passing blocks.
    for (int b = 0; b < NBLK; ++b) {
        hipMemsetAsync(agg, 0, (size_t)NN * HH * sizeof(float), stream);
        edge_block_kernel<<<EE / 64, 256, 0, stream>>>(
            edge_index, eBuf, hBuf, agg,
            f_bew1 + (size_t)b * 384 * HH, blk_edge_b1 + (size_t)b * HH,
            f_bew2 + (size_t)b * HH  * HH, blk_edge_b2 + (size_t)b * HH);
        node_block_kernel<<<NN / 16, 256, 0, stream>>>(
            hBuf, agg,
            f_bnw1 + (size_t)b * 256 * HH, blk_node_b1 + (size_t)b * HH,
            f_bnw2 + (size_t)b * HH  * HH, blk_node_b2 + (size_t)b * HH);
    }

    // Decoder.
    decoder_kernel<<<NN / 16, 256, 0, stream>>>(hBuf, f_dw1, dec_b1, dec_w2, dec_b2, (float*)d_out);
}
